// Attention_2D_55963423867340
// MI455X (gfx1250) — matrix operand order repaired and checked by reading, not compiled, not run on hardware
//
#include <hip/hip_runtime.h>
#include <hip/hip_bf16.h>

typedef _Float16 v8h  __attribute__((ext_vector_type(8)));
typedef _Float16 v16h __attribute__((ext_vector_type(16)));
typedef float    v8f  __attribute__((ext_vector_type(8)));

#define B_  8
#define L_  1024
#define C_  256
#define H_  8
#define D_  32
#define M_  (B_*L_)      /* 8192 rows */
#define KC_ (C_*9)       /* 2304 conv K */

__device__ __forceinline__ v8f zero_v8f() {
    v8f r;
#pragma unroll
    for (int i = 0; i < 8; ++i) r[i] = 0.f;
    return r;
}

// ---- DPP 16-lane butterfly reductions (no LDS traffic) ----
// ctrl: 0xB1 quad_perm[1,0,3,2] (xor1), 0x4E quad_perm[2,3,0,1] (xor2),
//       0x141 row_half_mirror (xor4 effective), 0x140 row_mirror (xor8 effective)
template<int CTRL>
__device__ __forceinline__ float dpp_bcast_f32(float x) {
    int v = __builtin_amdgcn_mov_dpp(__builtin_bit_cast(int, x), CTRL, 0xF, 0xF, true);
    return __builtin_bit_cast(float, v);
}
__device__ __forceinline__ float red_max16(float x) {
    x = fmaxf(x, dpp_bcast_f32<0xB1>(x));
    x = fmaxf(x, dpp_bcast_f32<0x4E>(x));
    x = fmaxf(x, dpp_bcast_f32<0x141>(x));
    x = fmaxf(x, dpp_bcast_f32<0x140>(x));
    return x;
}
__device__ __forceinline__ float red_add16(float x) {
    x += dpp_bcast_f32<0xB1>(x);
    x += dpp_bcast_f32<0x4E>(x);
    x += dpp_bcast_f32<0x141>(x);
    x += dpp_bcast_f32<0x140>(x);
    return x;
}

// A fragment: 16x32 f16 (MxK). lane<16: row=lane, K {0..7,16..23}; lane>=16: row=lane-16, K {8..15,24..31}
__device__ __forceinline__ v16h load_a_frag(const _Float16* A, int row0, int k0, int lda, int lane) {
    int m = lane & 15;
    const _Float16* p = A + (size_t)(row0 + m) * lda + k0 + ((lane & 16) ? 8 : 0);
    v8h lo = *(const v8h*)p;
    v8h hi = *(const v8h*)(p + 16);
    v16h r;
#pragma unroll
    for (int i = 0; i < 8; ++i) { r[i] = lo[i]; r[i + 8] = hi[i]; }
    return r;
}

// B fragment from B^T (N x K row-major): lane<16: col=n0+lane, K {0..7,16..23}; lane>=16: K {8..15,24..31}
__device__ __forceinline__ v16h load_b_frag(const _Float16* BT, int n0, int k0, int ldb, int lane) {
    int n = n0 + (lane & 15);
    const _Float16* p = BT + (size_t)n * ldb + k0 + ((lane & 16) ? 8 : 0);
    v8h lo = *(const v8h*)p;
    v8h hi = *(const v8h*)(p + 16);
    v16h r;
#pragma unroll
    for (int i = 0; i < 8; ++i) { r[i] = lo[i]; r[i + 8] = hi[i]; }
    return r;
}

__device__ __forceinline__ v8f wmma16(v16h a, v16h b, v8f c) {
    return __builtin_amdgcn_wmma_f32_16x16x32_f16(false, a, false, b, (short)0, c, false, false);
}

// ---------------- elementwise / prep kernels ----------------

__global__ void cast_f32_to_f16(const float* __restrict__ in, _Float16* __restrict__ out, int n) {
    int i = blockIdx.x * 256 + threadIdx.x;
    if (i < n) out[i] = (_Float16)in[i];
}

// im2col: src [B,L,C] channel-last -> dst [M_, KC_] f16, col = ci*9+ky*3+kx
__global__ void im2col_k(const float* __restrict__ src, _Float16* __restrict__ dst) {
    int idx = blockIdx.x * 256 + threadIdx.x;
    if (idx >= M_ * KC_) return;
    int row = idx / KC_;
    int kk  = idx % KC_;
    int ci = kk / 9; int r9 = kk % 9;
    int ky = r9 / 3; int kx = r9 % 3;
    int b = row >> 10; int p = row & 1023;
    int yy = (p >> 5) + ky - 1;
    int xx = (p & 31) + kx - 1;
    float v = 0.f;
    if ((unsigned)yy < 32u && (unsigned)xx < 32u)
        v = src[((size_t)b * L_ + yy * 32 + xx) * C_ + ci];
    dst[idx] = (_Float16)v;
}

// per-channel batch stats over all 8192 rows -> scale/shift for BN fold
__global__ void bn_stats(const float* __restrict__ conv, const float* __restrict__ gamma,
                         const float* __restrict__ beta, float* __restrict__ s, float* __restrict__ t) {
    int ch = blockIdx.x, tid = threadIdx.x;
    float sum = 0.f, sq = 0.f;
    for (int r = tid; r < M_; r += 256) {
        float v = conv[(size_t)r * C_ + ch];
        sum += v; sq += v * v;
    }
    __shared__ float s1[256], s2[256];
    s1[tid] = sum; s2[tid] = sq; __syncthreads();
    for (int off = 128; off; off >>= 1) {
        if (tid < off) { s1[tid] += s1[tid + off]; s2[tid] += s2[tid + off]; }
        __syncthreads();
    }
    if (tid == 0) {
        float mu  = s1[0] * (1.f / (float)M_);
        float var = s2[0] * (1.f / (float)M_) - mu * mu;
        float sc = gamma[ch] * rsqrtf(var + 1e-5f);
        s[ch] = sc;
        t[ch] = beta[ch] - mu * sc;
    }
}

// fold BN into projection: Weff[o,c] = W[o,c]*s[c] (f16); bias[o] = sum_c W[o,c]*t[c]
__global__ void fold_w(const float* __restrict__ W, const float* __restrict__ s, const float* __restrict__ t,
                       _Float16* __restrict__ Weff, float* __restrict__ bias) {
    int o = blockIdx.x, c = threadIdx.x;
    float w = W[(size_t)o * C_ + c];
    Weff[(size_t)o * C_ + c] = (_Float16)(w * s[c]);
    __shared__ float red[256];
    red[c] = w * t[c]; __syncthreads();
    for (int off = 128; off; off >>= 1) {
        if (c < off) red[c] += red[c + off];
        __syncthreads();
    }
    if (c == 0) bias[o] = red[0];
}

// v_proj [M_,C_] -> vT [B*H, 32, 1024]
__global__ void transpose_v(const _Float16* __restrict__ vproj, _Float16* __restrict__ vT) {
    int idx = blockIdx.x * 256 + threadIdx.x;  // M_*C_ threads
    int row = idx >> 8, c = idx & 255;
    int b = row >> 10, tt = row & 1023;
    int h = c >> 5,  d = c & 31;
    vT[(((size_t)(b * H_ + h)) * D_ + d) * L_ + tt] = vproj[idx];
}

// ---------------- WMMA GEMM: C = A(f16 MxK) * BT(f16 NxK)^T (+bias) ----------------
// block = 128 threads = 4 waves; block tile 128(M) x 64(N); wave tile 32x64
template<int OF32, int OF16, int BIAS>
__global__ void gemm_wmma(const _Float16* __restrict__ A, const _Float16* __restrict__ BT,
                          float* __restrict__ Cf, _Float16* __restrict__ Ch,
                          const float* __restrict__ bias,
                          int M, int N, int K, int lda, int ldb) {
    int lane = threadIdx.x & 31;
    int wave = threadIdx.x >> 5;
    int r0 = blockIdx.y * 128 + wave * 32;
    int c0 = blockIdx.x * 64;
    v8f acc[2][4];
#pragma unroll
    for (int a = 0; a < 2; ++a)
#pragma unroll
        for (int j = 0; j < 4; ++j) acc[a][j] = zero_v8f();

    for (int k0 = 0; k0 < K; k0 += 32) {
        v16h a0 = load_a_frag(A, r0,      k0, lda, lane);
        v16h a1 = load_a_frag(A, r0 + 16, k0, lda, lane);
#pragma unroll
        for (int j = 0; j < 4; ++j) {
            v16h b = load_b_frag(BT, c0 + 16 * j, k0, ldb, lane);
            acc[0][j] = wmma16(a0, b, acc[0][j]);
            acc[1][j] = wmma16(a1, b, acc[1][j]);
        }
    }
    int nb = lane & 15;
    int mh = (lane & 16) ? 8 : 0;
#pragma unroll
    for (int a = 0; a < 2; ++a)
#pragma unroll
        for (int j = 0; j < 4; ++j) {
            int n = c0 + 16 * j + nb;
            float bv = BIAS ? bias[n] : 0.f;
#pragma unroll
            for (int i = 0; i < 8; ++i) {
                int m = r0 + 16 * a + mh + i;
                float v = acc[a][j][i] + bv;
                if (OF32) Cf[(size_t)m * N + n] = v;
                if (OF16) Ch[(size_t)m * N + n] = (_Float16)v;
            }
        }
}

// ---------------- flash attention: one wave per (b,h, 16 query rows) ----------------
__global__ void attn_kernel(const _Float16* __restrict__ qproj, const _Float16* __restrict__ kproj,
                            const _Float16* __restrict__ vT, _Float16* __restrict__ outh) {
    const int bh = blockIdx.x;          // b*H + h
    const int b = bh >> 3, h = bh & 7;
    const int lane = threadIdx.x & 31;
    const int wave = threadIdx.x >> 5;
    const int q0 = blockIdx.y * 64 + wave * 16;

    __shared__ alignas(16) _Float16 ldsP[4][16 * 32];
    _Float16* P = ldsP[wave];

    const _Float16* qp = qproj + (size_t)b * L_ * C_ + h * D_;
    const _Float16* kp = kproj + (size_t)b * L_ * C_ + h * D_;
    const _Float16* vp = vT + (size_t)bh * D_ * L_;

    const int mh = (lane & 16) ? 8 : 0;
    const int nb = lane & 15;
    const float scale = 0.0625f;   // 256^-0.5

    v16h aq = load_a_frag(qp, q0, 0, C_, lane);   // 16x32, K = head dim
    v8f O0 = zero_v8f(), O1 = zero_v8f();
    float mrow[8], lrow[8];
#pragma unroll
    for (int i = 0; i < 8; ++i) { mrow[i] = -INFINITY; lrow[i] = 0.f; }

    for (int t0 = 0; t0 < L_; t0 += 32) {
        v16h bk0 = load_b_frag(kp, t0,      0, C_, lane);
        v16h bk1 = load_b_frag(kp, t0 + 16, 0, C_, lane);
        v8f s0 = wmma16(aq, bk0, zero_v8f());
        v8f s1 = wmma16(aq, bk1, zero_v8f());
#pragma unroll
        for (int i = 0; i < 8; ++i) {
            float a0 = s0[i] * scale;
            float a1 = s1[i] * scale;
            float mx = red_max16(fmaxf(a0, a1));          // DPP butterfly, VALU only
            float mnew = fmaxf(mrow[i], mx);
            float corr = __expf(mrow[i] - mnew);
            float p0 = __expf(a0 - mnew);
            float p1 = __expf(a1 - mnew);
            float rs = red_add16(p0 + p1);                // DPP butterfly, VALU only
            lrow[i] = lrow[i] * corr + rs;
            mrow[i] = mnew;
            O0[i] *= corr;
            O1[i] *= corr;
            P[(mh + i) * 32 + nb]      = (_Float16)p0;
            P[(mh + i) * 32 + 16 + nb] = (_Float16)p1;
        }
        asm volatile("s_wait_dscnt 0" ::: "memory");
        v16h ap  = load_a_frag(P, 0, 0, 32, lane);       // 16x32 P tile from LDS
        v16h bv0 = load_b_frag(vp, 0,  t0, L_, lane);    // dims 0..15
        v16h bv1 = load_b_frag(vp, 16, t0, L_, lane);    // dims 16..31
        O0 = wmma16(ap, bv0, O0);
        O1 = wmma16(ap, bv1, O1);
    }
#pragma unroll
    for (int i = 0; i < 8; ++i) {
        float inv = 1.f / lrow[i];
        size_t row = (size_t)(b * L_ + q0 + mh + i);
        outh[row * C_ + h * D_ + nb]      = (_Float16)(O0[i] * inv);
        outh[row * C_ + h * D_ + 16 + nb] = (_Float16)(O1[i] * inv);
    }
}

// ---------------- driver ----------------
extern "C" void kernel_launch(void* const* d_in, const int* in_sizes, int n_in,
                              void* d_out, int out_size, void* d_ws, size_t ws_size,
                              hipStream_t stream) {
    (void)in_sizes; (void)n_in; (void)out_size; (void)ws_size;
    const float* x       = (const float*)d_in[0];
    const float* y       = (const float*)d_in[1];
    const float* cqw     = (const float*)d_in[4];
    const float* bnqg    = (const float*)d_in[5];
    const float* bnqb    = (const float*)d_in[6];
    const float* ckw     = (const float*)d_in[7];
    const float* bnkg    = (const float*)d_in[8];
    const float* bnkb    = (const float*)d_in[9];
    const float* cvw     = (const float*)d_in[10];
    const float* bnvg    = (const float*)d_in[11];
    const float* bnvb    = (const float*)d_in[12];
    const float* Wq      = (const float*)d_in[13];
    const float* Wk      = (const float*)d_in[14];
    const float* Wv      = (const float*)d_in[15];
    const float* Wo      = (const float*)d_in[16];
    const float* bo      = (const float*)d_in[17];
    float* out = (float*)d_out;

    char* ws = (char*)d_ws;
    size_t off = 0;
    auto alloc = [&](size_t bytes) -> void* {
        void* p = ws + off;
        off += (bytes + 255) & ~(size_t)255;
        return p;
    };
    _Float16* colx    = (_Float16*)alloc((size_t)M_ * KC_ * 2);
    _Float16* coly    = (_Float16*)alloc((size_t)M_ * KC_ * 2);
    _Float16* wcq     = (_Float16*)alloc((size_t)C_ * KC_ * 2);
    _Float16* wck     = (_Float16*)alloc((size_t)C_ * KC_ * 2);
    _Float16* wcv     = (_Float16*)alloc((size_t)C_ * KC_ * 2);
    _Float16* Wo16    = (_Float16*)alloc((size_t)C_ * C_ * 2);
    float*    convf32 = (float*)   alloc((size_t)M_ * C_ * 4);   // reused for q,k,v
    _Float16* convq16 = (_Float16*)alloc((size_t)M_ * C_ * 2);
    _Float16* convk16 = (_Float16*)alloc((size_t)M_ * C_ * 2);
    _Float16* convv16 = (_Float16*)alloc((size_t)M_ * C_ * 2);
    float*    s_q = (float*)alloc(C_ * 4);  float* t_q = (float*)alloc(C_ * 4);
    float*    s_k = (float*)alloc(C_ * 4);  float* t_k = (float*)alloc(C_ * 4);
    float*    s_v = (float*)alloc(C_ * 4);  float* t_v = (float*)alloc(C_ * 4);
    _Float16* Weq = (_Float16*)alloc((size_t)C_ * C_ * 2);  float* biq = (float*)alloc(C_ * 4);
    _Float16* Wek = (_Float16*)alloc((size_t)C_ * C_ * 2);  float* bik = (float*)alloc(C_ * 4);
    _Float16* Wev = (_Float16*)alloc((size_t)C_ * C_ * 2);  float* biv = (float*)alloc(C_ * 4);
    _Float16* qproj = (_Float16*)alloc((size_t)M_ * C_ * 2);
    _Float16* kproj = (_Float16*)alloc((size_t)M_ * C_ * 2);
    _Float16* vproj = (_Float16*)alloc((size_t)M_ * C_ * 2);
    _Float16* vT    = (_Float16*)alloc((size_t)M_ * C_ * 2);
    _Float16* outh  = (_Float16*)alloc((size_t)M_ * C_ * 2);

    // weight casts
    cast_f32_to_f16<<<(C_ * KC_ + 255) / 256, 256, 0, stream>>>(cqw, wcq, C_ * KC_);
    cast_f32_to_f16<<<(C_ * KC_ + 255) / 256, 256, 0, stream>>>(ckw, wck, C_ * KC_);
    cast_f32_to_f16<<<(C_ * KC_ + 255) / 256, 256, 0, stream>>>(cvw, wcv, C_ * KC_);
    cast_f32_to_f16<<<(C_ * C_ + 255) / 256, 256, 0, stream>>>(Wo, Wo16, C_ * C_);

    // im2col
    im2col_k<<<(M_ * KC_ + 255) / 256, 256, 0, stream>>>(x, colx);
    im2col_k<<<(M_ * KC_ + 255) / 256, 256, 0, stream>>>(y, coly);

    dim3 gg(C_ / 64, M_ / 128);
    // conv GEMMs + stats (convf32 reused sequentially)
    gemm_wmma<1, 1, 0><<<gg, 128, 0, stream>>>(colx, wcq, convf32, convq16, nullptr, M_, C_, KC_, KC_, KC_);
    bn_stats<<<C_, 256, 0, stream>>>(convf32, bnqg, bnqb, s_q, t_q);
    gemm_wmma<1, 1, 0><<<gg, 128, 0, stream>>>(coly, wck, convf32, convk16, nullptr, M_, C_, KC_, KC_, KC_);
    bn_stats<<<C_, 256, 0, stream>>>(convf32, bnkg, bnkb, s_k, t_k);
    gemm_wmma<1, 1, 0><<<gg, 128, 0, stream>>>(coly, wcv, convf32, convv16, nullptr, M_, C_, KC_, KC_, KC_);
    bn_stats<<<C_, 256, 0, stream>>>(convf32, bnvg, bnvb, s_v, t_v);

    // fold BN into projections
    fold_w<<<C_, 256, 0, stream>>>(Wq, s_q, t_q, Weq, biq);
    fold_w<<<C_, 256, 0, stream>>>(Wk, s_k, t_k, Wek, bik);
    fold_w<<<C_, 256, 0, stream>>>(Wv, s_v, t_v, Wev, biv);

    // projections (f16 out)
    gemm_wmma<0, 1, 1><<<gg, 128, 0, stream>>>(convq16, Weq, nullptr, qproj, biq, M_, C_, C_, C_, C_);
    gemm_wmma<0, 1, 1><<<gg, 128, 0, stream>>>(convk16, Wek, nullptr, kproj, bik, M_, C_, C_, C_, C_);
    gemm_wmma<0, 1, 1><<<gg, 128, 0, stream>>>(convv16, Wev, nullptr, vproj, biv, M_, C_, C_, C_, C_);

    transpose_v<<<(M_ * C_) / 256, 256, 0, stream>>>(vproj, vT);

    attn_kernel<<<dim3(B_ * H_, L_ / 64), 128, 0, stream>>>(qproj, kproj, vT, outh);

    // output projection, f32 + bias -> d_out
    gemm_wmma<1, 0, 1><<<gg, 128, 0, stream>>>(outh, Wo16, out, nullptr, bo, M_, C_, C_, C_, C_);
}
